// MultiHeadAttention_65386582114369
// MI455X (gfx1250) — matrix operand order repaired and checked by reading, not compiled, not run on hardware
//
#include <hip/hip_runtime.h>
#include <hip/hip_bf16.h>

// ---------------------------------------------------------------------------
// MHA forward for B=4, S=2048, D=1024, H=16, HD=64 on gfx1250 (CDNA5, wave32)
// bf16 WMMA (v_wmma_f32_16x16x32_bf16) everywhere, fp32 accumulation.
// Softmax row reductions via DPP row_ror; K-tile staging via TDM
// (tensor_load_to_lds + s_wait_tensorcnt), 6-arg builtin form.
// ---------------------------------------------------------------------------

typedef __attribute__((ext_vector_type(16))) __bf16 v16bf;
typedef __attribute__((ext_vector_type(8)))  float  v8f;
typedef __attribute__((ext_vector_type(4)))  unsigned int u32x4;
typedef __attribute__((ext_vector_type(8)))  int  i32x8;
typedef __attribute__((ext_vector_type(4)))  int  i32x4;

#define BATCH  4
#define S_LEN  2048
#define D_DIM  1024
#define NHEAD  16
#define HDIM   64

#ifndef __has_builtin
#define __has_builtin(x) 0
#endif
#if __has_builtin(__builtin_amdgcn_tensor_load_to_lds) && \
    __has_builtin(__builtin_amdgcn_s_wait_tensorcnt)
#define USE_TDM 1
#else
#define USE_TDM 0
#endif

union Frag {
    v16bf  v;
    uint4  u[2];
    __bf16 h[16];
};

__device__ __forceinline__ uint4 ldg16(const __bf16* p) {
    return *reinterpret_cast<const uint4*>(p);
}

// DPP row rotate (within each 16-lane row of the wave32). CTRL = 0x120 | n.
template <int CTRL>
__device__ __forceinline__ float dpp_rot(float x) {
    int xi = __float_as_int(x);
    int yi = __builtin_amdgcn_update_dpp(xi, xi, CTRL, 0xf, 0xf, true);
    return __int_as_float(yi);
}
// Max / sum over the 16 lanes of a half-wave (circular rotate reduction).
__device__ __forceinline__ float row16_max(float x) {
    x = fmaxf(x, dpp_rot<0x121>(x));
    x = fmaxf(x, dpp_rot<0x122>(x));
    x = fmaxf(x, dpp_rot<0x124>(x));
    x = fmaxf(x, dpp_rot<0x128>(x));
    return x;
}
__device__ __forceinline__ float row16_sum(float x) {
    x += dpp_rot<0x121>(x);
    x += dpp_rot<0x122>(x);
    x += dpp_rot<0x124>(x);
    x += dpp_rot<0x128>(x);
    return x;
}

// ---------------------------------------------------------------------------
// fp32 -> bf16 elementwise conversion
// ---------------------------------------------------------------------------
__global__ void f32_to_bf16_kernel(const float* __restrict__ in,
                                   __bf16* __restrict__ out, int n) {
    int i = (blockIdx.x * blockDim.x + threadIdx.x) * 4;
    if (i + 3 < n) {
        float4 f = *reinterpret_cast<const float4*>(in + i);
        out[i + 0] = (__bf16)f.x;
        out[i + 1] = (__bf16)f.y;
        out[i + 2] = (__bf16)f.z;
        out[i + 3] = (__bf16)f.w;
    }
}

// ---------------------------------------------------------------------------
// C[M,N] = A[M,K] @ W[N,K]^T + bias   (torch Linear semantics)
// block = 128 threads (4 waves); block tile 128x64; wave tile 32x64.
// Two A fragments (rows r..r+15, r+16..r+31) share each B fragment:
// 12 vector loads per 8 WMMAs.
// ---------------------------------------------------------------------------
template <bool OUT_BF16>
__global__ void gemm_bias_kernel(const __bf16* __restrict__ A,
                                 const __bf16* __restrict__ W,
                                 const float*  __restrict__ bias,
                                 __bf16* __restrict__ outb,
                                 float*  __restrict__ outf,
                                 int M, int N, int K) {
    const int lane = threadIdx.x & 31;
    const int wave = threadIdx.x >> 5;
    const int half = lane >> 4;   // 0: lanes 0-15, 1: lanes 16-31
    const int l16  = lane & 15;

    const int row0 = blockIdx.y * 128 + wave * 32;
    const int col0 = blockIdx.x * 64;

    v8f acc[2][4];
#pragma unroll
    for (int rh = 0; rh < 2; ++rh)
#pragma unroll
        for (int c = 0; c < 4; ++c)
#pragma unroll
            for (int i = 0; i < 8; ++i) acc[rh][c][i] = 0.0f;

    const __bf16* arow0 = A + (size_t)(row0 + l16) * K;
    const __bf16* arow1 = A + (size_t)(row0 + 16 + l16) * K;
    const __bf16* wbase = W + (size_t)(col0 + l16) * K;

    for (int k = 0; k < K; k += 32) {
        if (k + 64 < K) {
            __builtin_prefetch(arow0 + k + 64, 0, 1);
            __builtin_prefetch(arow1 + k + 64, 0, 1);
            __builtin_prefetch(wbase + k + 64, 0, 1);
        }
        Frag a0, a1;
        a0.u[0] = ldg16(arow0 + k +     half * 16);
        a0.u[1] = ldg16(arow0 + k + 8 + half * 16);
        a1.u[0] = ldg16(arow1 + k +     half * 16);
        a1.u[1] = ldg16(arow1 + k + 8 + half * 16);
        const int boff = k + half * 16;
#pragma unroll
        for (int c = 0; c < 4; ++c) {
            const __bf16* wp = wbase + (size_t)c * 16 * K + boff;
            Frag b;
            b.u[0] = ldg16(wp);
            b.u[1] = ldg16(wp + 8);
            acc[0][c] = __builtin_amdgcn_wmma_f32_16x16x32_bf16(
                false, a0.v, false, b.v, (short)0, acc[0][c], false, false);
            acc[1][c] = __builtin_amdgcn_wmma_f32_16x16x32_bf16(
                false, a1.v, false, b.v, (short)0, acc[1][c], false, false);
        }
    }

#pragma unroll
    for (int c = 0; c < 4; ++c) {
        const int col = col0 + c * 16 + l16;
        const float bv = bias[col];
#pragma unroll
        for (int rh = 0; rh < 2; ++rh) {
#pragma unroll
            for (int v = 0; v < 8; ++v) {
                const int r = row0 + rh * 16 + (half ? v + 8 : v);
                const float val = acc[rh][c][v] + bv;
                if constexpr (OUT_BF16)
                    outb[(size_t)r * N + col] = (__bf16)val;
                else
                    outf[(size_t)r * N + col] = val;
            }
        }
    }
}

// ---------------------------------------------------------------------------
// Fused causal flash attention. grid = (S/64, B*H), block = 128 (4 waves).
// Wave w owns query rows [qblk*64 + w*16, +16). Block streams 32-key tiles.
// K tile staged by the Tensor Data Mover (pad feature reproduces the 72-elem
// padded LDS stride); V tile transposed cooperatively.
// ---------------------------------------------------------------------------
#define KLDS_ELEMS (32 * 72)   // offset 0 in the shared block
#define VT_ELEMS   (64 * 40)
#define P_ELEMS    (4 * 16 * 40)

__global__ void flash_attn_kernel(const __bf16* __restrict__ Qb,
                                  const __bf16* __restrict__ Kb,
                                  const __bf16* __restrict__ Vb,
                                  __bf16* __restrict__ Cb) {
    __shared__ __align__(16) __bf16 smem[KLDS_ELEMS + VT_ELEMS + P_ELEMS];
    __bf16* Klds = smem;                         // [key][hd], stride 72
    __bf16* Vt   = smem + KLDS_ELEMS;            // [hd][key], stride 40
    __bf16* Plds = smem + KLDS_ELEMS + VT_ELEMS; // per-wave P, stride 40

    const int tid  = threadIdx.x;
    const int lane = tid & 31;
    const int wave = tid >> 5;
    const int half = lane >> 4;
    const int l16  = lane & 15;

    const int b      = blockIdx.y >> 4;
    const int h      = blockIdx.y & 15;
    const int qblk   = blockIdx.x;
    const int seq_r0 = qblk * 64 + wave * 16;

    // Q A-operands for this wave's 16 rows: hd chunks [0,32) and [32,64)
    Frag qa0, qa1;
    {
        const __bf16* qrow =
            Qb + (size_t)(b * S_LEN + seq_r0 + l16) * D_DIM + h * HDIM;
        qa0.u[0] = ldg16(qrow +      half * 16);
        qa0.u[1] = ldg16(qrow + 8  + half * 16);
        qa1.u[0] = ldg16(qrow + 32 + half * 16);
        qa1.u[1] = ldg16(qrow + 40 + half * 16);
    }

    v8f ctx[4];
#pragma unroll
    for (int c = 0; c < 4; ++c)
#pragma unroll
        for (int i = 0; i < 8; ++i) ctx[c][i] = 0.0f;

    float mrow[8], lrow[8];
#pragma unroll
    for (int v = 0; v < 8; ++v) { mrow[v] = -3.0e38f; lrow[v] = 0.0f; }

    const int   nkb   = 2 * qblk + 2;   // causal: keys up to qblk*64+63
    const float scale = 0.125f;         // 1/sqrt(64)

    for (int kb = 0; kb < nkb; ++kb) {
        const int key0 = kb * 32;
        __syncthreads();  // previous iteration's LDS reads done

#if USE_TDM
        // --- K tile via Tensor Data Mover (one wave issues the descriptor) --
        if (wave == 0) {
            const unsigned long long ga = (unsigned long long)(const void*)
                (Kb + (size_t)(b * S_LEN + key0) * D_DIM + h * HDIM);
            // D# group 0: count=1, lds_addr=0 (Klds), 57-bit global addr,
            // type=2 ("image").
            u32x4 g0;
            g0[0] = 1u;
            g0[1] = 0u;
            g0[2] = (unsigned)(ga & 0xFFFFFFFFu);
            g0[3] = (unsigned)((ga >> 32) & 0x1FFFFFFu) | (2u << 30);
            // D# group 1: data_size=2B, pad_enable, pad every 32 DWORDs
            // (128 B row) by 4 DWORDs (16 B)  ->  LDS row stride 72 elems.
            // tensor_dim0=64, tensor_dim1=2048, tile 64x32, dim0 stride 1024.
            i32x8 g1;
            g1[0] = (1 << 16) | (1 << 20) | (4 << 22) | (3 << 25);
            g1[1] = (64 << 16);            // tensor_dim0 = 64
            g1[2] = (int)(2048u << 16);    // tensor_dim1 = 2048
            g1[3] = (64 << 16);            // tile_dim0 = 64
            g1[4] = 32;                    // tile_dim1 = 32
            g1[5] = 1024;                  // tensor_dim0_stride = D
            g1[6] = 0;
            g1[7] = 0;
            const i32x4 gz4 = {0, 0, 0, 0};          // groups 2/3: 2D tensor
            const i32x8 gz8 = {0, 0, 0, 0, 0, 0, 0, 0};
            __builtin_amdgcn_tensor_load_to_lds(g0, g1, gz4, gz4, gz8, 0);
        }
#endif
        // --- V tile transposed [hd][key] (and K tile without TDM) ----------
        {
            const int key = tid >> 2;   // 0..31
            const int ch  = tid & 3;    // 16 hd elems each
            const size_t grow = (size_t)(b * S_LEN + key0 + key) * D_DIM +
                                h * HDIM + ch * 16;
#if !USE_TDM
            uint4 k0 = ldg16(Kb + grow);
            uint4 k1 = ldg16(Kb + grow + 8);
            *reinterpret_cast<uint4*>(&Klds[key * 72 + ch * 16])     = k0;
            *reinterpret_cast<uint4*>(&Klds[key * 72 + ch * 16 + 8]) = k1;
#endif
            Frag vf;
            vf.u[0] = ldg16(Vb + grow);
            vf.u[1] = ldg16(Vb + grow + 8);
#pragma unroll
            for (int i = 0; i < 16; ++i)
                Vt[(ch * 16 + i) * 40 + key] = vf.h[i];
        }
#if USE_TDM
        if (wave == 0) __builtin_amdgcn_s_wait_tensorcnt(0);
#endif
        __syncthreads();

        // Scores: S = Q @ K^T, two 16-key sub-tiles, K-dim = HD = 64 (2 wmma)
        v8f s[2];
#pragma unroll
        for (int j = 0; j < 2; ++j) {
            const __bf16* kp = &Klds[(j * 16 + l16) * 72 + half * 16];
            Frag b0, b1;
            b0.u[0] = *reinterpret_cast<const uint4*>(kp);
            b0.u[1] = *reinterpret_cast<const uint4*>(kp + 8);
            b1.u[0] = *reinterpret_cast<const uint4*>(kp + 32);
            b1.u[1] = *reinterpret_cast<const uint4*>(kp + 40);
            v8f z;
#pragma unroll
            for (int i = 0; i < 8; ++i) z[i] = 0.0f;
            z = __builtin_amdgcn_wmma_f32_16x16x32_bf16(
                false, qa0.v, false, b0.v, (short)0, z, false, false);
            z = __builtin_amdgcn_wmma_f32_16x16x32_bf16(
                false, qa1.v, false, b1.v, (short)0, z, false, false);
            s[j] = z;
        }

        // Online softmax (fp32). Row stats per half-wave (matches C layout).
        float p0[8], p1[8], alpha[8];
#pragma unroll
        for (int v = 0; v < 8; ++v) {
            const int r     = seq_r0 + (half ? v + 8 : v);
            const int key_a = key0 + l16;
            const int key_b = key0 + 16 + l16;
            float sa = s[0][v] * scale; if (key_a > r) sa = -3.0e38f;
            float sb = s[1][v] * scale; if (key_b > r) sb = -3.0e38f;
            const float mt   = row16_max(fmaxf(sa, sb));
            const float mnew = fmaxf(mrow[v], mt);
            alpha[v] = __expf(mrow[v] - mnew);
            p0[v]    = __expf(sa - mnew);
            p1[v]    = __expf(sb - mnew);
            const float rs = row16_sum(p0[v] + p1[v]);
            lrow[v] = lrow[v] * alpha[v] + rs;
            mrow[v] = mnew;
        }
#pragma unroll
        for (int c = 0; c < 4; ++c)
#pragma unroll
            for (int v = 0; v < 8; ++v) ctx[c][v] *= alpha[v];

        // Stage P (16x32 bf16) through LDS into A-operand layout
        {
            __bf16* pw = &Plds[wave * 640];
#pragma unroll
            for (int v = 0; v < 8; ++v) {
                const int rl = half ? v + 8 : v;
                pw[rl * 40 + l16]      = (__bf16)p0[v];
                pw[rl * 40 + 16 + l16] = (__bf16)p1[v];
            }
        }
        __syncthreads();

        Frag pa;
        {
            const __bf16* pr = &Plds[wave * 640 + l16 * 40];
            pa.u[0] = *reinterpret_cast<const uint4*>(pr +     half * 16);
            pa.u[1] = *reinterpret_cast<const uint4*>(pr + 8 + half * 16);
        }
        // ctx += P @ V  (K-dim = 32 keys, 4 wmma over HD columns)
#pragma unroll
        for (int c = 0; c < 4; ++c) {
            const __bf16* vp = &Vt[(c * 16 + l16) * 40 + half * 16];
            Frag vb;
            vb.u[0] = *reinterpret_cast<const uint4*>(vp);
            vb.u[1] = *reinterpret_cast<const uint4*>(vp + 8);
            ctx[c] = __builtin_amdgcn_wmma_f32_16x16x32_bf16(
                false, pa.v, false, vb.v, (short)0, ctx[c], false, false);
        }
    }

    // Normalize and write ctx (bf16) in [B*S, D] layout
#pragma unroll
    for (int v = 0; v < 8; ++v) {
        const float inv = 1.0f / lrow[v];
        const int r = seq_r0 + (half ? v + 8 : v);
        const size_t base = (size_t)(b * S_LEN + r) * D_DIM + h * HDIM;
#pragma unroll
        for (int c = 0; c < 4; ++c)
            Cb[base + c * 16 + l16] = (__bf16)(ctx[c][v] * inv);
    }
}

// ---------------------------------------------------------------------------
// Host launcher
// ---------------------------------------------------------------------------
extern "C" void kernel_launch(void* const* d_in, const int* in_sizes, int n_in,
                              void* d_out, int out_size, void* d_ws,
                              size_t ws_size, hipStream_t stream) {
    (void)in_sizes; (void)n_in; (void)out_size; (void)ws_size;

    const float* x  = (const float*)d_in[0];
    // d_in[1] is the causal mask -> applied analytically in flash_attn_kernel
    const float* Wq = (const float*)d_in[2];
    const float* bq = (const float*)d_in[3];
    const float* Wk = (const float*)d_in[4];
    const float* bk = (const float*)d_in[5];
    const float* Wv = (const float*)d_in[6];
    const float* bv = (const float*)d_in[7];
    const float* Wo = (const float*)d_in[8];
    const float* bo = (const float*)d_in[9];
    float* out = (float*)d_out;

    char* ws = (char*)d_ws;
    size_t off = 0;
    auto take = [&](size_t bytes) -> char* {
        char* p = ws + off;
        off += (bytes + 255) & ~(size_t)255;
        return p;
    };

    const size_t MD = (size_t)BATCH * S_LEN * D_DIM;  // 8,388,608
    const size_t DD = (size_t)D_DIM * D_DIM;          // 1,048,576

    __bf16* xb  = (__bf16*)take(MD * 2);
    __bf16* wqb = (__bf16*)take(DD * 2);
    __bf16* wkb = (__bf16*)take(DD * 2);
    __bf16* wvb = (__bf16*)take(DD * 2);
    __bf16* wob = (__bf16*)take(DD * 2);
    __bf16* qb  = (__bf16*)take(MD * 2);
    __bf16* kb  = (__bf16*)take(MD * 2);
    __bf16* vb2 = (__bf16*)take(MD * 2);
    __bf16* cb  = (__bf16*)take(MD * 2);

    const int CT = 256;
    f32_to_bf16_kernel<<<dim3((unsigned)(MD / 4 / CT)), CT, 0, stream>>>(x,  xb,  (int)MD);
    f32_to_bf16_kernel<<<dim3((unsigned)(DD / 4 / CT)), CT, 0, stream>>>(Wq, wqb, (int)DD);
    f32_to_bf16_kernel<<<dim3((unsigned)(DD / 4 / CT)), CT, 0, stream>>>(Wk, wkb, (int)DD);
    f32_to_bf16_kernel<<<dim3((unsigned)(DD / 4 / CT)), CT, 0, stream>>>(Wv, wvb, (int)DD);
    f32_to_bf16_kernel<<<dim3((unsigned)(DD / 4 / CT)), CT, 0, stream>>>(Wo, wob, (int)DD);

    const int M = BATCH * S_LEN;  // 8192
    dim3 gg(D_DIM / 64, M / 128); // (16, 64)
    gemm_bias_kernel<true><<<gg, 128, 0, stream>>>(xb, wqb, bq, qb,  nullptr, M, D_DIM, D_DIM);
    gemm_bias_kernel<true><<<gg, 128, 0, stream>>>(xb, wkb, bk, kb,  nullptr, M, D_DIM, D_DIM);
    gemm_bias_kernel<true><<<gg, 128, 0, stream>>>(xb, wvb, bv, vb2, nullptr, M, D_DIM, D_DIM);

    flash_attn_kernel<<<dim3(S_LEN / 64, BATCH * NHEAD), 128, 0, stream>>>(qb, kb, vb2, cb);

    gemm_bias_kernel<false><<<gg, 128, 0, stream>>>(cb, wob, bo, nullptr, out, M, D_DIM, D_DIM);
}
